// PseudoProjector_38843684225318
// MI455X (gfx1250) — matrix operand order repaired and checked by reading, not compiled, not run on hardware
//
#include <hip/hip_runtime.h>
#include <hip/hip_bf16.h>

#define DIM    4096
#define DC     512
#define NROWS  16384      // 4 * 4096
#define EPSV   1e-10f

typedef __bf16 bf16_t;
typedef __attribute__((ext_vector_type(16))) __bf16 v16bf;
typedef __attribute__((ext_vector_type(8)))  __bf16 v8bf;
typedef __attribute__((ext_vector_type(8)))  float  v8f;

__device__ __forceinline__ bf16_t f2bf(float f) {
  union { float f; unsigned u; } c; c.f = f;
  unsigned u = c.u;
  // round-to-nearest-even truncation to bf16
  unsigned r = (u + 0x7FFFu + ((u >> 16) & 1u)) >> 16;
  union { unsigned short s; bf16_t b; } o; o.s = (unsigned short)r;
  return o.b;
}

// ---------------- conversion kernels ----------------

__global__ void cvt_f32_bf16(const float* __restrict__ src, bf16_t* __restrict__ dst, int n) {
  int i = blockIdx.x * blockDim.x + threadIdx.x;
  int stride = gridDim.x * blockDim.x;
  for (; i < n; i += stride) dst[i] = f2bf(src[i]);
}

// src [rows][cols] f32 -> dst [cols][rows] bf16
__global__ void cvt_transpose_bf16(const float* __restrict__ src, bf16_t* __restrict__ dst,
                                   int rows, int cols) {
  int i = blockIdx.x * blockDim.x + threadIdx.x;
  int n = rows * cols;
  int stride = gridDim.x * blockDim.x;
  for (; i < n; i += stride) {
    int r = i / cols, c = i - r * cols;
    dst[(size_t)c * rows + r] = f2bf(src[i]);
  }
}

// ---------------- build augmented [M^T | I] ----------------
// M^T = T0 + I + eps  (eps added to every entry, matching reference)
__global__ void build_aug(const float* __restrict__ T0, float* __restrict__ Aug) {
  int i = blockIdx.x * blockDim.x + threadIdx.x;
  int stride = gridDim.x * blockDim.x;
  const int total = DC * 2 * DC;
  for (; i < total; i += stride) {
    int r = i >> 10;          // row (2*DC == 1024 columns)
    int c = i & 1023;
    float v;
    if (c < DC) v = T0[r * DC + c] + ((r == c) ? 1.0f : 0.0f) + EPSV;
    else        v = ((c - DC) == r) ? 1.0f : 0.0f;
    Aug[i] = v;
  }
}

// ---------------- Gauss-Jordan inverse (single workgroup) ----------------
// Inverts the left half of Aug (= M^T); writes inv(M^T) = inv(M)^T row-major.
__global__ void __launch_bounds__(1024) gauss_jordan(float* __restrict__ Aug,
                                                     float* __restrict__ MinvT) {
  __shared__ float prow[2 * DC];
  __shared__ float fac[DC];
  __shared__ float pivinv;
  const int tid = threadIdx.x;
  for (int k = 0; k < DC; ++k) {
    if (tid == 0) pivinv = 1.0f / Aug[(size_t)k * 2 * DC + k];
    __syncthreads();
    float pi = pivinv;
    for (int j = tid; j < 2 * DC; j += 1024) {
      float v = Aug[(size_t)k * 2 * DC + j] * pi;
      Aug[(size_t)k * 2 * DC + j] = v;
      prow[j] = v;
    }
    __syncthreads();
    for (int i = tid; i < DC; i += 1024)
      fac[i] = (i == k) ? 0.0f : Aug[(size_t)i * 2 * DC + k];
    __syncthreads();
    for (int idx = tid; idx < DC * 2 * DC; idx += 1024) {
      int i = idx >> 10;
      if (i != k) {
        int j = idx & (2 * DC - 1);
        Aug[idx] -= fac[i] * prow[j];
      }
    }
    __syncthreads();
  }
  for (int idx = tid; idx < DC * DC; idx += 1024) {
    int i = idx >> 9, j = idx & (DC - 1);
    MinvT[idx] = Aug[(size_t)i * 2 * DC + DC + j];
  }
}

// ---------------- WMMA bf16 NT GEMM ----------------
// D[M,N] = A[M,K] * W[N,K]^T ; A,W bf16 row-major; D f32 or bf16.
// Fragment layouts per CDNA5 ISA 7.12.2 (16-bit A 16x32, B 32x16).
// Each wave owns a 32x64 output tile: 2 A-frags x 4 B-frags -> 8 WMMAs
// per 12 b128 loads per K-step of 32.

__device__ __forceinline__ v16bf load_a_frag(const bf16_t* __restrict__ A, int lda,
                                             int row0, int k0, int lane) {
  int m = row0 + (lane & 15);
  int half = lane >> 4;
  const bf16_t* p = A + (size_t)m * lda + k0 + 8 * half;
  v8bf lo = *(const v8bf*)(p);        // k = k0+8h    .. +7  (slots 0..7)
  v8bf hi = *(const v8bf*)(p + 16);   // k = k0+16+8h .. +7  (slots 8..15)
  v16bf a;
#pragma unroll
  for (int i = 0; i < 8; ++i) { a[i] = lo[i]; a[i + 8] = hi[i]; }
  return a;
}

__device__ __forceinline__ v16bf load_b_frag_nt(const bf16_t* __restrict__ W, int ldw,
                                                int col0, int k0, int lane) {
  int n = col0 + (lane & 15);
  int half = lane >> 4;
  const bf16_t* p = W + (size_t)n * ldw + k0 + 8 * half;  // k = k0+8h..+7 (slots 0..7), k0+16+8h..+7 (slots 8..15)
  v8bf lo = *(const v8bf*)(p);
  v8bf hi = *(const v8bf*)(p + 16);
  v16bf b;
#pragma unroll
  for (int i = 0; i < 8; ++i) { b[i] = lo[i]; b[i + 8] = hi[i]; }
  return b;
}

template <bool OUT_BF16>
__global__ void __launch_bounds__(256) gemm_nt_wmma(const bf16_t* __restrict__ A, int lda,
                                                    const bf16_t* __restrict__ W, int ldw,
                                                    void* __restrict__ Dv, int N, int K) {
  const int lane = threadIdx.x & 31;
  const int wave = threadIdx.x >> 5;
  const int row0 = blockIdx.y * 32;                  // 32 rows per wave tile
  const int colbase = blockIdx.x * 512 + wave * 64;  // 64 cols per wave tile

  v8f acc[2][4] = {};
  for (int k0 = 0; k0 < K; k0 += 32) {
    v16bf a0 = load_a_frag(A, lda, row0,      k0, lane);
    v16bf a1 = load_a_frag(A, lda, row0 + 16, k0, lane);
#pragma unroll
    for (int t = 0; t < 4; ++t) {
      v16bf b = load_b_frag_nt(W, ldw, colbase + t * 16, k0, lane);
      acc[0][t] = __builtin_amdgcn_wmma_f32_16x16x32_bf16(false, a0, false, b, (short)0,
                                                          acc[0][t], false, false);
      acc[1][t] = __builtin_amdgcn_wmma_f32_16x16x32_bf16(false, a1, false, b, (short)0,
                                                          acc[1][t], false, false);
    }
  }

  const int half = lane >> 4;
  const int nlo = lane & 15;
#pragma unroll
  for (int s = 0; s < 2; ++s) {
#pragma unroll
    for (int t = 0; t < 4; ++t) {
      int n = colbase + t * 16 + nlo;
#pragma unroll
      for (int r = 0; r < 8; ++r) {
        size_t off = (size_t)(row0 + s * 16 + r + 8 * half) * N + n;
        if (OUT_BF16) ((bf16_t*)Dv)[off] = f2bf(acc[s][t][r]);
        else          ((float*)Dv)[off]  = acc[s][t][r];
      }
    }
  }
}

// ---------------- host launcher ----------------

extern "C" void kernel_launch(void* const* d_in, const int* in_sizes, int n_in,
                              void* d_out, int out_size, void* d_ws, size_t ws_size,
                              hipStream_t stream) {
  (void)in_sizes; (void)n_in; (void)out_size; (void)ws_size;
  const float* features   = (const float*)d_in[0];  // [4,4096,4096] -> [16384,4096]
  const float* restrict_w = (const float*)d_in[1];  // [512,4096]
  const float* prolong_w  = (const float*)d_in[2];  // [4096,512]
  float* out = (float*)d_out;                       // [16384,4096]

  char* ws = (char*)d_ws;
  size_t off = 0;
  auto alloc = [&](size_t bytes) -> void* {
    void* p = ws + off;
    off += (bytes + 255) & ~(size_t)255;
    return p;
  };

  bf16_t* x_bf     = (bf16_t*)alloc((size_t)NROWS * DIM * 2);  // 128 MB
  bf16_t* R_bf     = (bf16_t*)alloc((size_t)DC * DIM * 2);     // 4 MB
  bf16_t* P_bf     = (bf16_t*)alloc((size_t)DIM * DC * 2);     // 4 MB
  bf16_t* PT_bf    = (bf16_t*)alloc((size_t)DC * DIM * 2);     // 4 MB
  float*  T0       = (float*) alloc((size_t)DC * DC * 4);      // 1 MB
  float*  Aug      = (float*) alloc((size_t)DC * 2 * DC * 4);  // 2 MB
  float*  MinvT_f  = (float*) alloc((size_t)DC * DC * 4);      // 1 MB
  bf16_t* MinvT_bf = (bf16_t*)alloc((size_t)DC * DC * 2);      // 0.5 MB
  bf16_t* c_bf     = (bf16_t*)alloc((size_t)NROWS * DC * 2);   // 16 MB
  bf16_t* u_bf     = (bf16_t*)alloc((size_t)NROWS * DC * 2);   // 16 MB

  // 1) precision conversion (memory-bound, ~0.5 GB total traffic)
  cvt_f32_bf16<<<4096, 256, 0, stream>>>(features, x_bf, NROWS * DIM);
  cvt_f32_bf16<<<1024, 256, 0, stream>>>(restrict_w, R_bf, DC * DIM);
  cvt_f32_bf16<<<1024, 256, 0, stream>>>(prolong_w, P_bf, DIM * DC);
  cvt_transpose_bf16<<<1024, 256, 0, stream>>>(prolong_w, PT_bf, DIM, DC);

  // 2) T0 = restrict_w @ prolong_w  (NT via pre-transposed prolong): [512,512]
  gemm_nt_wmma<false><<<dim3(1, DC / 32), 256, 0, stream>>>(R_bf, DIM, PT_bf, DIM, T0, DC, DIM);

  // 3) build [M^T | I] and invert -> inv(M)^T (row-major) for NT-consumption
  build_aug<<<512, 256, 0, stream>>>(T0, Aug);
  gauss_jordan<<<1, 1024, 0, stream>>>(Aug, MinvT_f);
  cvt_f32_bf16<<<256, 256, 0, stream>>>(MinvT_f, MinvT_bf, DC * DC);

  // 4) coarsened = x @ restrict_w^T : [16384,512], bf16 out
  gemm_nt_wmma<true><<<dim3(1, NROWS / 32), 256, 0, stream>>>(x_bf, DIM, R_bf, DIM, c_bf, DC, DIM);

  // 5) u = coarsened @ inv(M)  (NT with W = inv(M)^T) : [16384,512], bf16 out
  gemm_nt_wmma<true><<<dim3(1, NROWS / 32), 256, 0, stream>>>(c_bf, DC, MinvT_bf, DC, u_bf, DC, DC);

  // 6) projected = u @ prolong_w^T : [16384,4096], f32 out
  gemm_nt_wmma<false><<<dim3(DIM / 512, NROWS / 32), 256, 0, stream>>>(u_bf, DC, P_bf, DC, out, DIM, DC);
}
